// TemporalSelfAttentionClusterKey_39702677684477
// MI455X (gfx1250) — matrix operand order repaired and checked by reading, not compiled, not run on hardware
//
#include <hip/hip_runtime.h>
#include <stdint.h>

typedef __attribute__((ext_vector_type(16))) __bf16          v16bf;
typedef __attribute__((ext_vector_type(8)))  __bf16          v8bf;
typedef __attribute__((ext_vector_type(8)))  float           v8f;
typedef __attribute__((ext_vector_type(8)))  float           v8fv;
typedef __attribute__((ext_vector_type(8)))  unsigned short  v8us;
typedef __attribute__((ext_vector_type(16))) unsigned short  v16us;

#define B_  8
#define T_  192
#define N_  64
#define D_  512
#define H_  8
#define DH_ 64
#define C_  8

// padded LDS pitches (ushorts): per-lane word stride gcd(stride,64)==4 -> no
// bank conflicts for 16B ds loads; pitch stays a multiple of 8 (16B aligned)
#define KP_ 72     // K tile rows of 64
#define VP_ 200    // V^T tile rows of 192
#define PP_ 200    // P strip rows of 192

// ---------- helpers ----------------------------------------------------------

__device__ __forceinline__ unsigned short f2bf(float f) {
  __bf16 h = (__bf16)f;                       // native v_cvt(_pk)_bf16_f32
  return __builtin_bit_cast(unsigned short, h);
}

__device__ __forceinline__ v8f wmma_bf16(v16bf a, v16bf b, v8f c) {
  return __builtin_amdgcn_wmma_f32_16x16x32_bf16(false, a, false, b, (short)0, c,
                                                 false, false);
}

// A fragment (16x32 bf16) from an fp32 row: two 32B loads + packed cvt
__device__ __forceinline__ v16bf ldA_f32(const float* row, int kb, int lh) {
  v8fv f0 = *(const v8fv*)(row + kb + 16 * lh);
  v8fv f1 = *(const v8fv*)(row + kb + 16 * lh + 8);
  union { v16bf v; v8bf h[2]; } u;
  u.h[0] = __builtin_convertvector(f0, v8bf);
  u.h[1] = __builtin_convertvector(f1, v8bf);
  return u.v;
}

// A fragment (16x32 bf16) from bf16 storage (global or LDS)
__device__ __forceinline__ v16bf ldA_bf(const unsigned short* row, int kb, int lh) {
  union { v16bf v; v8us h[2]; } u;
  u.h[0] = *(const v8us*)(row + kb + 16 * lh);
  u.h[1] = *(const v8us*)(row + kb + 16 * lh + 8);
  return u.v;
}

// B fragment (32x16 bf16), global transposed storage: one 32B load
__device__ __forceinline__ v16bf ldB_bf(const unsigned short* row, int kb, int lh) {
  return *(const v16bf*)(row + kb + 16 * lh);
}

// B fragment from LDS (rows only 16B aligned due to padding): two 16B loads
__device__ __forceinline__ v16bf ldB_lds(const unsigned short* row, int kb, int lh) {
  const unsigned short* p = row + kb + 16 * lh;
  union { v16bf v; v8us h[2]; } u;
  u.h[0] = *(const v8us*)p;
  u.h[1] = *(const v8us*)(p + 8);
  return u.v;
}

// CDNA5 async DMA: 16B global -> LDS, tracked by ASYNCcnt
__device__ __forceinline__ void async_ld16(unsigned lds_off, const void* gaddr) {
  asm volatile("global_load_async_to_lds_b128 %0, %1, off"
               : : "v"(lds_off), "v"(gaddr) : "memory");
}
__device__ __forceinline__ void wait_async0() {
  asm volatile("s_wait_asynccnt 0" ::: "memory");
}

// ---------- kernel 0: weight convert / transpose to bf16 ---------------------

__global__ __launch_bounds__(256) void prep_weights(
    const float* __restrict__ Wq, const float* __restrict__ Wv,
    const float* __restrict__ Pj, const float* __restrict__ Wk,
    unsigned short* __restrict__ WqB, unsigned short* __restrict__ WvB,
    unsigned short* __restrict__ PjB, unsigned short* __restrict__ WkT) {
  const int S = D_ * D_;                         // 262144
  int idx = blockIdx.x * 256 + threadIdx.x;      // grid covers 11*S
  if (idx < S) {
    WqB[idx] = f2bf(Wq[idx]);
  } else if (idx < 2 * S) {
    WvB[idx - S] = f2bf(Wv[idx - S]);
  } else if (idx < 3 * S) {
    PjB[idx - 2 * S] = f2bf(Pj[idx - 2 * S]);
  } else {
    int t = idx - 3 * S;
    int c = t >> 18;
    int r = t & (S - 1);
    int e = r >> 9, d = r & 511;
    WkT[(size_t)c * S + (size_t)e * D_ + d] =
        f2bf(Wk[(size_t)c * S + (size_t)d * D_ + e]);
  }
}

// ---------- kernel 1: fused grouped QKV projection ---------------------------

__global__ __launch_bounds__(256) void qkv_gemm(
    const float* __restrict__ x,
    const unsigned short* __restrict__ WqB,
    const unsigned short* __restrict__ WvB,
    const unsigned short* __restrict__ WkT,
    const float* __restrict__ Wq_b, const float* __restrict__ Wv_b,
    const float* __restrict__ bk, const int* __restrict__ cid,
    unsigned short* __restrict__ Qb, unsigned short* __restrict__ Kb,
    unsigned short* __restrict__ Vt) {
  __shared__ unsigned short vs[8][DH_][24];      // 24 KB, padded (stride 12 words)
  const int lane = threadIdx.x & 31;
  const int wave = threadIdx.x >> 5;
  const int ll = lane & 15, lh = lane >> 4;
  const int eblk = blockIdx.x;                   // == head index h
  const int m0 = blockIdx.y * 128 + wave * 16;   // row in [0,1536) = b*T+t
  const int e0 = eblk * 64;
  const int n  = blockIdx.z;
  const int c  = cid[n];

  const unsigned short* WkC = WkT + (size_t)c * (D_ * D_);
  const float* arow = x + ((size_t)(m0 + ll) * N_ + n) * D_;

  v8f zero = {0.f, 0.f, 0.f, 0.f, 0.f, 0.f, 0.f, 0.f};
  v8f aq[4], ak[4], av[4];
#pragma unroll
  for (int j = 0; j < 4; ++j) { aq[j] = zero; ak[j] = zero; av[j] = zero; }

  for (int kb = 0; kb < D_; kb += 32) {
    v16bf a = ldA_f32(arow, kb, lh);
#pragma unroll
    for (int j = 0; j < 4; ++j) {
      size_t ro = (size_t)(e0 + 16 * j + ll) * D_;
      aq[j] = wmma_bf16(a, ldB_bf(WqB + ro, kb, lh), aq[j]);
      ak[j] = wmma_bf16(a, ldB_bf(WkC + ro, kb, lh), ak[j]);
      av[j] = wmma_bf16(a, ldB_bf(WvB + ro, kb, lh), av[j]);
    }
  }

  const int bI = m0 / T_;                        // strip never crosses batch
  const int t0 = m0 - bI * T_;
  const size_t head = ((size_t)bI * N_ + n) * H_ + eblk;   // h == eblk
  unsigned short* qdst = Qb + (head * T_ + t0) * DH_;
  unsigned short* kdst = Kb + (head * T_ + t0) * DH_;

#pragma unroll
  for (int j = 0; j < 4; ++j) {
    const int e = e0 + 16 * j + ll;
    const int dh = 16 * j + ll;
    const float bq = Wq_b[e];
    const float bkv = bk[(size_t)c * D_ + e];
    const float bv = Wv_b[e];
#pragma unroll
    for (int r = 0; r < 8; ++r) {
      const int tl = r + 8 * lh;                 // local row 0..15
      qdst[(size_t)tl * DH_ + dh] = f2bf(aq[j][r] + bq);
      kdst[(size_t)tl * DH_ + dh] = f2bf(ak[j][r] + bkv);
      vs[wave][dh][tl] = f2bf(av[j][r] + bv);    // LDS transpose staging
    }
  }
  // wave-private LDS transpose -> coalesced 32B stores of V^T rows
  {
    const int dh0 = lane;                        // 0..31
    union { v16us w; v8us h[2]; } r0u, r1u;
    r0u.h[0] = *(const v8us*)&vs[wave][dh0][0];
    r0u.h[1] = *(const v8us*)&vs[wave][dh0][8];
    r1u.h[0] = *(const v8us*)&vs[wave][dh0 + 32][0];
    r1u.h[1] = *(const v8us*)&vs[wave][dh0 + 32][8];
    *(v16us*)(Vt + (head * DH_ + dh0) * T_ + t0) = r0u.w;
    *(v16us*)(Vt + (head * DH_ + dh0 + 32) * T_ + t0) = r1u.w;
  }
}

// ---------- kernel 2: attention per head -------------------------------------
// 4096 blocks, 128 thr = 4 waves. K & V tiles async-DMA'd once into padded LDS
// (ASYNCcnt + barrier), then 12 strip-GEMMs read LDS conflict-free.

__global__ __launch_bounds__(128) void attn_kernel(
    const unsigned short* __restrict__ Qb, const unsigned short* __restrict__ Kb,
    const unsigned short* __restrict__ Vt, unsigned short* __restrict__ Oh) {
  extern __shared__ unsigned short smem[];
  unsigned short* Ks = smem;                        // [192][KP_]  27648 B
  unsigned short* Vs = Ks + T_ * KP_;               // [64][VP_]   25600 B
  unsigned short* Pb = Vs + DH_ * VP_;              // [4][16][PP_] 25600 B

  const int lane = threadIdx.x & 31, wave = threadIdx.x >> 5;
  const int ll = lane & 15, lh = lane >> 4;
  const int head = blockIdx.x;                   // (b*64+n)*8+h
  const int h = head & 7;
  const int n = (head >> 3) & 63;
  const int b = head >> 9;
  const unsigned short* Qh = Qb + (size_t)head * (T_ * DH_);
  const unsigned short* Kh = Kb + (size_t)head * (T_ * DH_);
  const unsigned short* Vh = Vt + (size_t)head * (DH_ * T_);
  const float sc_log2e = 0.125f * 1.44269504088896f;   // (1/sqrt(64))*log2(e)
  v8f zero = {0.f, 0.f, 0.f, 0.f, 0.f, 0.f, 0.f, 0.f};

  // ---- async stage K (192x64) and V^T (64x192) into padded LDS ----
  {
    const unsigned ldsK = (unsigned)(size_t)(const void*)Ks;
    const unsigned ldsV = (unsigned)(size_t)(const void*)Vs;
    const char* gK = (const char*)Kh;
    const char* gV = (const char*)Vh;
#pragma unroll
    for (int i = 0; i < 12; ++i) {                  // 1536 16B chunks, 8/row
      int cch = threadIdx.x + 128 * i;
      int r = cch >> 3, cc = cch & 7;
      async_ld16(ldsK + r * (KP_ * 2) + cc * 16, gK + cch * 16);
    }
#pragma unroll
    for (int i = 0; i < 12; ++i) {                  // 1536 16B chunks, 24/row
      int cch = threadIdx.x + 128 * i;
      int r = cch / 24, cc = cch - r * 24;
      async_ld16(ldsV + r * (VP_ * 2) + cc * 16, gV + cch * 16);
    }
    wait_async0();
    __syncthreads();
  }

  unsigned short* Pw = Pb + wave * 16 * PP_;

  for (int st = 0; st < 3; ++st) {
    const int r0 = (wave * 3 + st) * 16;
    // ---- scores strip: S[16 x 192] = Q_strip · K^T (K from LDS) ----
    v8f s[12];
#pragma unroll
    for (int j = 0; j < 12; ++j) s[j] = zero;
    const unsigned short* qrow = Qh + (size_t)(r0 + ll) * DH_;
#pragma unroll
    for (int kb = 0; kb < DH_; kb += 32) {
      v16bf a = ldA_bf(qrow, kb, lh);
#pragma unroll
      for (int j = 0; j < 12; ++j) {
        v16bf kf = ldB_lds(Ks + (16 * j + ll) * KP_, kb, lh);
        s[j] = wmma_bf16(a, kf, s[j]);
      }
    }
    // ---- softmax (unnormalized); rows split across 16-lane halves ----
    float sum[8];
#pragma unroll
    for (int e = 0; e < 8; ++e) {
      float m = s[0][e];
#pragma unroll
      for (int j = 1; j < 12; ++j) m = fmaxf(m, s[j][e]);
      m = fmaxf(m, __shfl_xor(m, 1));
      m = fmaxf(m, __shfl_xor(m, 2));
      m = fmaxf(m, __shfl_xor(m, 4));
      m = fmaxf(m, __shfl_xor(m, 8));
      float sm = 0.f;
#pragma unroll
      for (int j = 0; j < 12; ++j) {
        float p = exp2f((s[j][e] - m) * sc_log2e);
        s[j][e] = p;
        sm += p;
      }
      sm += __shfl_xor(sm, 1);
      sm += __shfl_xor(sm, 2);
      sm += __shfl_xor(sm, 4);
      sm += __shfl_xor(sm, 8);
      sum[e] = sm;
    }
    // ---- stash P strip (bf16) in wave-private padded LDS ----
#pragma unroll
    for (int e = 0; e < 8; ++e) {
      int rl = e + 8 * lh;
#pragma unroll
      for (int j = 0; j < 12; ++j) Pw[rl * PP_ + 16 * j + ll] = f2bf(s[j][e]);
    }
    // ---- out strip: O[16 x 64] = P · V  (both operands in LDS) ----
    v8f o[4];
#pragma unroll
    for (int j = 0; j < 4; ++j) o[j] = zero;
    const unsigned short* prow = Pw + ll * PP_;
#pragma unroll
    for (int kb = 0; kb < T_; kb += 32) {
      v16bf a = ldA_bf(prow, kb, lh);
#pragma unroll
      for (int j = 0; j < 4; ++j) {
        v16bf vf = ldB_lds(Vs + (16 * j + ll) * VP_, kb, lh);
        o[j] = wmma_bf16(a, vf, o[j]);
      }
    }
    // ---- normalize by 1/rowsum & store ----
    size_t obase = (((size_t)b * T_ + r0) * N_ + n) * D_ + h * DH_;
#pragma unroll
    for (int r = 0; r < 8; ++r) {
      float inv = 1.0f / sum[r];
      size_t rowoff = obase + (size_t)(r + 8 * lh) * (N_ * D_);
#pragma unroll
      for (int j = 0; j < 4; ++j) Oh[rowoff + 16 * j + ll] = f2bf(o[j][r] * inv);
    }
  }
}

// ---------- kernel 3: output projection (fp32 out) ---------------------------

__global__ __launch_bounds__(256) void proj_gemm(
    const unsigned short* __restrict__ A, const unsigned short* __restrict__ Wp,
    const float* __restrict__ pb, float* __restrict__ out) {
  const int lane = threadIdx.x & 31, wave = threadIdx.x >> 5;
  const int ll = lane & 15, lh = lane >> 4;
  const int e0 = blockIdx.x * 64;
  const int m0 = blockIdx.y * 128 + wave * 16;   // (b*T+t)*N+n flat row
  const unsigned short* arow = A + (size_t)(m0 + ll) * D_;
  v8f zero = {0.f, 0.f, 0.f, 0.f, 0.f, 0.f, 0.f, 0.f};
  v8f acc[4];
#pragma unroll
  for (int j = 0; j < 4; ++j) acc[j] = zero;

  for (int kb = 0; kb < D_; kb += 32) {
    v16bf a = ldA_bf(arow, kb, lh);
#pragma unroll
    for (int j = 0; j < 4; ++j) {
      v16bf b = ldB_bf(Wp + (size_t)(e0 + 16 * j + ll) * D_, kb, lh);
      acc[j] = wmma_bf16(a, b, acc[j]);
    }
  }
  float* obase = out + (size_t)m0 * D_ + e0;
#pragma unroll
  for (int j = 0; j < 4; ++j) {
    int e = e0 + 16 * j + ll;
    float bv = pb[e];
#pragma unroll
    for (int r = 0; r < 8; ++r)
      obase[(size_t)(r + 8 * lh) * D_ + 16 * j + ll] = acc[j][r] + bv;
  }
}

// ---------- launcher ---------------------------------------------------------

extern "C" void kernel_launch(void* const* d_in, const int* in_sizes, int n_in,
                              void* d_out, int out_size, void* d_ws, size_t ws_size,
                              hipStream_t stream) {
  (void)in_sizes; (void)n_in; (void)out_size; (void)ws_size;
  const float* x      = (const float*)d_in[0];
  const float* Wq_w   = (const float*)d_in[1];
  const float* Wq_b   = (const float*)d_in[2];
  const float* Wv_w   = (const float*)d_in[3];
  const float* Wv_b   = (const float*)d_in[4];
  const float* Wk     = (const float*)d_in[5];
  const float* bk     = (const float*)d_in[6];
  const float* proj_w = (const float*)d_in[7];
  const float* proj_b = (const float*)d_in[8];
  const int*   cid    = (const int*)d_in[9];
  float* out = (float*)d_out;

  char* ws = (char*)d_ws;
  const size_t SQ = (size_t)4096 * T_ * DH_ * 2;     // 100,663,296 B per tensor
  unsigned short* Qb  = (unsigned short*)(ws);
  unsigned short* Kb  = (unsigned short*)(ws + SQ);
  unsigned short* Vt  = (unsigned short*)(ws + 2 * SQ);
  unsigned short* Oh  = (unsigned short*)(ws + 3 * SQ);
  unsigned short* WqB = (unsigned short*)(ws + 4 * SQ);
  unsigned short* WvB = WqB + (size_t)D_ * D_;
  unsigned short* PjB = WvB + (size_t)D_ * D_;
  unsigned short* WkT = PjB + (size_t)D_ * D_;

  prep_weights<<<11264, 256, 0, stream>>>(Wq_w, Wv_w, proj_w, Wk, WqB, WvB, PjB, WkT);
  qkv_gemm<<<dim3(8, 12, 64), 256, 0, stream>>>(x, WqB, WvB, WkT, Wq_b, Wv_b,
                                                bk, cid, Qb, Kb, Vt);
  const int attn_lds = (T_ * KP_ + DH_ * VP_ + 4 * 16 * PP_) * 2;   // 78848 B
  attn_kernel<<<4096, 128, attn_lds, stream>>>(Qb, Kb, Vt, Oh);
  proj_gemm<<<dim3(8, 768), 256, 0, stream>>>(Oh, PjB, proj_b, out);
}
